// Chromatin_Network_14731737825686
// MI455X (gfx1250) — matrix operand order repaired and checked by reading, not compiled, not run on hardware
//
#include <hip/hip_runtime.h>

typedef __attribute__((ext_vector_type(16))) _Float16 v16h;
typedef __attribute__((ext_vector_type(8)))  _Float16 v8h;
typedef __attribute__((ext_vector_type(8)))  float    v8f;

#define LYR 30
#define HID 30
#define TT  500
#define NT  8      // n-tiles: 4H=120 padded to 128
#define KT  2      // k-tiles: 2H=60 padded to 64 (two K=32 WMMAs)

// ---------------- prep: repack weights into WMMA B-fragment layout -----------
// wpack[l][kt][nt][lane(32)][half(16)], f16.
// B[k][n] = Wc[nglob][kglob];  lane = n + 16*((k>>3)&1), half = (k&7) + 8*(k>=16).
// K layout: kg in [0,30)  -> input weights (layer0: only kg==0 = w_ih0)
//           kg == 30      -> fused bias column (b_ih+b_hh), A supplies 1.0 there
//           kg in [32,62) -> hidden weights
__global__ void pack_weights_kernel(const float* __restrict__ w_ih0,
                                    const float* __restrict__ w_ih,
                                    const float* __restrict__ w_hh,
                                    const float* __restrict__ b_ih,
                                    const float* __restrict__ b_hh,
                                    _Float16* __restrict__ wpack, int total) {
    int idx = blockIdx.x * blockDim.x + threadIdx.x;
    if (idx >= total) return;
    int hh   = idx & 15;
    int lane = (idx >> 4) & 31;
    int nt   = (idx >> 9) & 7;
    int kt   = (idx >> 12) & 1;
    int l    = idx >> 13;
    int n  = lane & 15;
    int k  = (hh < 8) ? ((lane >> 4) << 3) + hh : 16 + ((lane >> 4) << 3) + (hh - 8);   // 0..31 within K-tile
    int kg = kt * 32 + k;               // 0..63
    int ng = nt * 16 + n;               // 0..127
    float v = 0.0f;
    if (ng < 4 * HID) {
        if (kg < HID) {                 // input-weight part
            v = (l == 0) ? (kg == 0 ? w_ih0[ng] : 0.0f)
                         : w_ih[((l - 1) * 4 * HID + ng) * HID + kg];
        } else if (kg == HID) {         // fused bias column (kg==30, kt==0 only)
            v = b_ih[l * 4 * HID + ng] + b_hh[l * 4 * HID + ng];
        } else if (kg >= 32 && kg < 32 + HID) {   // hidden-weight part
            v = w_hh[(l * 4 * HID + ng) * HID + (kg - 32)];
        }
    }
    wpack[idx] = (_Float16)v;
}

// ---------------- device helpers ---------------------------------------------
__device__ __forceinline__ float fsig(float x)  { return 1.0f / (1.0f + __expf(-x)); }
__device__ __forceinline__ float ftanh(float x) {
    x = fminf(fmaxf(x, -15.0f), 15.0f);
    float e = __expf(2.0f * x);
    return (e - 1.0f) / (e + 1.0f);
}
__device__ __forceinline__ v16h zero_v16h() {
    v16h v;
#pragma unroll
    for (int i = 0; i < 16; ++i) v[i] = (_Float16)0.0f;
    return v;
}
__device__ __forceinline__ v8f zero_v8f() {
    v8f v;
#pragma unroll
    for (int i = 0; i < 8; ++i) v[i] = 0.0f;
    return v;
}

// ---------------- fused LSTM + head ------------------------------------------
// One wave (32 threads) per workgroup; each owns 16 batch rows through all
// T=500 timesteps and L=30 layers. h kept in LDS in A-fragment layout, with a
// constant 1.0 baked into the k=30 slot (lane>=16, half 14) feeding the fused
// bias column. Weight B-fragments are software-pipelined: layer l+1's 16
// fragments load while layer l's activation phase runs.
__global__ __launch_bounds__(32) void lstm_fused_kernel(
    const float* __restrict__ x, const _Float16* __restrict__ wpack,
    const float* __restrict__ w1, const float* __restrict__ b1,
    const float* __restrict__ w2, const float* __restrict__ b2,
    const float* __restrict__ w3, const float* __restrict__ b3,
    const float* __restrict__ w4, const float* __restrict__ b4,
    float* __restrict__ out)
{
    // LDS budget: 30720 + 28800 + 3840 + 1920 = 65280 B  (<= 64 KB)
    __shared__ __align__(32) _Float16 hfrag[LYR][32][16];   // h in A-frag layout
    __shared__ _Float16 cst[LYR][16 * HID];                 // cell state
    __shared__ __align__(32) _Float16 gst[4 * HID][16];     // gates, [col][row]
    __shared__ float    hout[16][HID];                      // final top-layer h

    const int lane = threadIdx.x;
    const int m0   = blockIdx.x * 16;

    for (int i = lane; i < LYR * 32 * 16; i += 32) ((_Float16*)hfrag)[i] = (_Float16)0.0f;
    for (int i = lane; i < LYR * 16 * HID; i += 32) ((_Float16*)cst)[i] = (_Float16)0.0f;
    if (lane >= 16)                                        // constant-1 feature (k=30)
        for (int l = 0; l < LYR; ++l) hfrag[l][lane][14] = (_Float16)1.0f;
    __syncthreads();

    const v16h* wv = (const v16h*)wpack;   // [l][kt][nt][lane] -> 32B per lane

    // pipeline prologue: layer 0 weight fragments in registers
    v16h b0f[NT], b1f[NT];
#pragma unroll
    for (int nt = 0; nt < NT; ++nt) {
        b0f[nt] = wv[(0 * NT + nt) * 32 + lane];
        b1f[nt] = wv[(1 * NT + nt) * 32 + lane];
    }

    for (int t = 0; t < TT; ++t) {
        // layer-0 input fragment: A[m][0] = x_t[m], A[m][30] = 1.0, rest 0
        v16h inp = zero_v16h();
        if (lane < 16) inp[0]  = (_Float16)x[(m0 + lane) * TT + t];
        else           inp[14] = (_Float16)1.0f;

        for (int l = 0; l < LYR; ++l) {
            v16h ah = *(const v16h*)&hfrag[l][lane][0];   // hidden fragment (k 32..63)

            v8f acc[NT];
#pragma unroll
            for (int nt = 0; nt < NT; ++nt) {
                acc[nt] = __builtin_amdgcn_wmma_f32_16x16x32_f16(
                    false, inp, false, b0f[nt], (short)0, zero_v8f(), false, false);
                acc[nt] = __builtin_amdgcn_wmma_f32_16x16x32_f16(
                    false, ah,  false, b1f[nt], (short)0, acc[nt], false, false);
            }

            // software pipeline: start next layer's weight loads now; they
            // complete in the shadow of the gates spill + activation phase.
            const int ln = (l + 1 == LYR) ? 0 : l + 1;
#pragma unroll
            for (int nt = 0; nt < NT; ++nt) {
                b0f[nt] = wv[((ln * KT + 0) * NT + nt) * 32 + lane];
                b1f[nt] = wv[((ln * KT + 1) * NT + nt) * 32 + lane];
            }

            // spill gates: lane holds one column x 8 rows -> one 16B LDS store
            const int col0  = lane & 15;
            const int rbase = (lane < 16) ? 0 : 8;
#pragma unroll
            for (int nt = 0; nt < NT; ++nt) {
                int col = nt * 16 + col0;
                if (col < 4 * HID) {
                    v8h hv;
#pragma unroll
                    for (int r = 0; r < 8; ++r) hv[r] = (_Float16)acc[nt][r];
                    *(v8h*)&gst[col][rbase] = hv;
                }
            }
            __syncthreads();

            // elementwise LSTM cell: 480 elements, 15 per lane
            for (int e = lane; e < 16 * HID; e += 32) {
                int m = e / HID, j = e % HID;
                float gi = fsig((float)gst[j][m]);
                float gf = fsig((float)gst[HID + j][m]);
                float gg = ftanh((float)gst[2 * HID + j][m]);
                float go = fsig((float)gst[3 * HID + j][m]);
                float cold = (float)cst[l][m * HID + j];
                float cn = gf * cold + gi * gg;
                float hn = go * ftanh(cn);
                cst[l][m * HID + j] = (_Float16)cn;
                // scatter h into A-fragment layout (16-bit A table, ISA 7.12.2)
                int tl, hh;
                if      (j <  8) { tl = m;      hh = j;      }
                else if (j < 16) { tl = m + 16; hh = j - 8;  }
                else if (j < 24) { tl = m;      hh = j - 8;  }
                else             { tl = m + 16; hh = j - 16; }
                hfrag[l][tl][hh] = (_Float16)hn;
                if (l == LYR - 1 && t == TT - 1) hout[m][j] = hn;
            }
            __syncthreads();
            // this layer's fresh h (incl. baked-in 1.0) is next layer's input
            inp = *(const v16h*)&hfrag[l][lane][0];
        }
    }
    __syncthreads();

    // MLP head 30->25->10->5->1, softmax over size-1 axis, round
    if (lane < 16) {
        int m = lane;
        float a1[25];
        for (int i = 0; i < 25; ++i) {
            float s = b1[i];
            for (int j = 0; j < HID; ++j) s += w1[i * HID + j] * hout[m][j];
            a1[i] = fmaxf(s, 0.0f);
        }
        float a2[10];
        for (int i = 0; i < 10; ++i) {
            float s = b2[i];
            for (int j = 0; j < 25; ++j) s += w2[i * 25 + j] * a1[j];
            a2[i] = fmaxf(s, 0.0f);
        }
        float a3[5];
        for (int i = 0; i < 5; ++i) {
            float s = b3[i];
            for (int j = 0; j < 10; ++j) s += w3[i * 10 + j] * a2[j];
            a3[i] = fmaxf(s, 0.0f);
        }
        float z = b4[0];
        for (int j = 0; j < 5; ++j) z += w4[j] * a3[j];
        float p = __expf(z - z);          // softmax over a size-1 axis == 1.0
        out[m0 + m] = rintf(p);           // jnp.round (nearest-even)
    }
}

// ---------------- launch ------------------------------------------------------
extern "C" void kernel_launch(void* const* d_in, const int* in_sizes, int n_in,
                              void* d_out, int out_size, void* d_ws, size_t ws_size,
                              hipStream_t stream) {
    const float* x     = (const float*)d_in[0];
    const float* w_ih0 = (const float*)d_in[1];
    const float* w_ih  = (const float*)d_in[2];
    const float* w_hh  = (const float*)d_in[3];
    const float* b_ih  = (const float*)d_in[4];
    const float* b_hh  = (const float*)d_in[5];
    const float* w1 = (const float*)d_in[6];  const float* b1 = (const float*)d_in[7];
    const float* w2 = (const float*)d_in[8];  const float* b2 = (const float*)d_in[9];
    const float* w3 = (const float*)d_in[10]; const float* b3 = (const float*)d_in[11];
    const float* w4 = (const float*)d_in[12]; const float* b4 = (const float*)d_in[13];
    float* out = (float*)d_out;

    const int B = in_sizes[0] / TT;               // 16384

    // workspace layout: [wpack f16, 480KB]
    _Float16* wpack = (_Float16*)d_ws;
    const int wtotal = LYR * KT * NT * 32 * 16;   // 245760

    pack_weights_kernel<<<(wtotal + 255) / 256, 256, 0, stream>>>(
        w_ih0, w_ih, w_hh, b_ih, b_hh, wpack, wtotal);

    lstm_fused_kernel<<<B / 16, 32, 0, stream>>>(
        x, wpack, w1, b1, w2, b2, w3, b3, w4, b4, out);
}
